// TransformerTranducer_15891378995674
// MI455X (gfx1250) — matrix operand order repaired and checked by reading, not compiled, not run on hardware
//
#include <hip/hip_runtime.h>
#include <hip/hip_bf16.h>
#include <stdint.h>

// ---------------------------------------------------------------------------
// Transformer Transducer forward (RNNT loss) for MI455X / gfx1250.
//
// Precision strategy: all GEMMs via v_wmma_f32_16x16x32_f16 (f16 in, f32
// accumulate).  The joint network is fused: tanh(a+l) tile -> WMMA against
// f16-transposed dense_w -> streaming logsumexp -> blank/target gather, so
// the 134MB [B,T,U,V] log_probs tensor never exists in HBM.
// The joint B-tile (32KB/step from L2-resident dense_w) is moved by the
// Tensor Data Mover (tensor_load_to_lds + s_wait_tensorcnt).
// ---------------------------------------------------------------------------

#define B_  4
#define T_  256
#define U_  64
#define H_  512
#define NH_ 8
#define DH_ 64
#define F_  2048
#define V_  512
#define NEGI (-1.0e30f)

typedef _Float16 half_t;
typedef half_t v16h __attribute__((ext_vector_type(16)));
typedef half_t v8h  __attribute__((ext_vector_type(8)));
typedef float  v8f  __attribute__((ext_vector_type(8)));
typedef unsigned int u32x4 __attribute__((ext_vector_type(4)));
typedef int i32x4 __attribute__((ext_vector_type(4)));
typedef int i32x8 __attribute__((ext_vector_type(8)));

// ---- WMMA fragment helpers (this code's k-contiguous order, not the ISA table order; wave32) ----------
// A (16x32 f16): lanes 0-15 -> M=lane, K = 0..15; lanes 16-31 ->
// M=lane-16, K = 16..31.  With k-contiguous LDS rows this is one
// contiguous 16-half (32B) load.
__device__ __forceinline__ v16h make_a_frag(const half_t* base, int strideH,
                                            int lane, int kOff) {
  const int m  = lane & 15;
  const int kb = kOff + ((lane >> 4) << 4);
  const half_t* p = base + m * strideH + kb;
  return *(const v16h*)p;
}

// B (32x16 f16): lanes 0-15 -> N=lane, K=0..15; lanes 16-31 -> N=lane-16,
// K=16..31.  Staged n-major/k-contiguous in LDS -> one 32B v16h load.
__device__ __forceinline__ v16h make_b_frag(const half_t* base, int strideH,
                                            int lane, int nOff, int kOff) {
  const int n  = nOff + (lane & 15);
  const int kb = kOff + ((lane >> 4) << 4);
  return *(const v16h*)(base + n * strideH + kb);
}

__device__ __forceinline__ v8f wmma_f16(v16h a, v16h b, v8f c) {
  return __builtin_amdgcn_wmma_f32_16x16x32_f16(false, a, false, b,
                                                (short)0, c, false, false);
}

// ---- Tensor Data Mover: DMA a 2D tile (tileX elems x tileY rows, rowStride
// elems, 2-byte data) from global into LDS.  D# layout per ISA ch.8.
__device__ __forceinline__ void tdm_load_tile_f16(
    unsigned ldsAddr, const half_t* gsrc, unsigned tensorX, unsigned tensorY,
    unsigned tileX, unsigned tileY, unsigned rowStride) {
  const unsigned long long ga = (unsigned long long)(uintptr_t)gsrc;
  u32x4 g0;
  g0[0] = 1u;                                     // count=1, user mode
  g0[1] = ldsAddr;                                // lds_addr [63:32]
  g0[2] = (unsigned)(ga & 0xFFFFFFFFull);         // global_addr [95:64]
  g0[3] = (unsigned)((ga >> 32) & 0x01FFFFFFull)  // global_addr [120:96]
        | (2u << 30);                             // type = 2 ("image")
  i32x8 g1;
  g1[0] = (int)(1u << 16);                        // data_size = 1 -> 2 bytes
  g1[1] = (int)((tensorX & 0xFFFFu) << 16);       // tensor_dim0[15:0]
  g1[2] = (int)((tensorX >> 16) & 0xFFFFu) | (int)((tensorY & 0xFFFFu) << 16);
  g1[3] = (int)((tensorY >> 16) & 0xFFFFu) | (int)(tileX << 16);  // tile_dim0
  g1[4] = (int)tileY;                             // tile_dim1 (tile_dim2 = 0)
  g1[5] = (int)rowStride;                         // tensor_dim0_stride lo32
  g1[6] = 0;
  g1[7] = 0;
  i32x4 gz = {0, 0, 0, 0};
#if __clang_major__ >= 23
  i32x8 gz8 = {0, 0, 0, 0, 0, 0, 0, 0};
  __builtin_amdgcn_tensor_load_to_lds(g0, g1, gz, gz, gz8, 0);
#else
  __builtin_amdgcn_tensor_load_to_lds(g0, g1, gz, gz, 0);
#endif
}

// ---------------------------------------------------------------------------
// K1: x[b,t,:] = audio[b,:,t] + audio_pos[t,:]
__global__ __launch_bounds__(256) void audio_pos_kernel(
    const float* __restrict__ audio, const float* __restrict__ pos,
    float* __restrict__ x) {
  const int row = blockIdx.x;            // b*T + t
  const int b = row >> 8, t = row & 255;
  const int tid = threadIdx.x;
#pragma unroll
  for (int i = 0; i < 2; ++i) {
    const int c = tid + i * 256;
    x[(long)row * H_ + c] = audio[(long)b * H_ * T_ + (long)c * T_ + t]
                          + pos[t * H_ + c];
  }
}

// K2: y[b,u,:] = word_emb[label[b,u],:] + label_pos[u,:]
__global__ __launch_bounds__(256) void embed_kernel(
    const int* __restrict__ labels, const float* __restrict__ emb,
    const float* __restrict__ pos, float* __restrict__ y) {
  const int row = blockIdx.x;            // b*U + u
  const int u = row & (U_ - 1);
  const int tid = threadIdx.x;
  const int idx = labels[row];
#pragma unroll
  for (int i = 0; i < 2; ++i) {
    const int c = tid + i * 256;
    y[(long)row * H_ + c] = emb[(long)idx * H_ + c] + pos[u * H_ + c];
  }
}

// ---------------------------------------------------------------------------
// K3: generic WMMA GEMM:  C[M,N] = act(A[M,K] @ W[K,N] + bias)
// Block tile 64x128 (8 waves = 4M x 2N, each wave 16x64 = 4 accum tiles).
template <int ACT>   // 0 = none, 1 = relu
__global__ __launch_bounds__(256) void gemm_kernel(
    const float* __restrict__ A, const float* __restrict__ W,
    const float* __restrict__ bias, float* __restrict__ C,
    int M, int N, int K) {
  __shared__ __align__(64) half_t As[64][32];
  __shared__ __align__(64) half_t Bst[128][32];   // n-major, k contiguous
  const int tid = threadIdx.x, lane = tid & 31, wv = tid >> 5;
  const int mw = wv & 3, nw = wv >> 2;
  const int mBase = blockIdx.y * 64;
  const int nBase = blockIdx.x * 128;
  v8f acc[4] = {};
  for (int kk = 0; kk < K; kk += 32) {
    __syncthreads();
    if (kk + 32 < K) {   // prefetch next A/B chunks -> global_prefetch_b8
      __builtin_prefetch(&A[(long)(mBase + (tid >> 2)) * K + kk + 32 + (tid & 3) * 8], 0, 1);
      __builtin_prefetch(&W[(long)(kk + 32 + (tid >> 3)) * N + nBase + (tid & 7) * 16], 0, 1);
    }
    for (int e = tid; e < 64 * 32; e += 256) {
      const int r = e >> 5, c = e & 31;
      As[r][c] = (half_t)A[(long)(mBase + r) * K + kk + c];
    }
    for (int e = tid; e < 32 * 128; e += 256) {
      const int r = e >> 7, c = e & 127;
      Bst[c][r] = (half_t)W[(long)(kk + r) * N + nBase + c];
    }
    __syncthreads();
    const v16h af = make_a_frag(&As[mw * 16][0], 32, lane, 0);
#pragma unroll
    for (int t = 0; t < 4; ++t) {
      const v16h bf = make_b_frag(&Bst[0][0], 32, lane, nw * 64 + t * 16, 0);
      acc[t] = wmma_f16(af, bf, acc[t]);
    }
  }
  // C/D layout: VGPR r -> M = r (lanes 0-15) / r+8 (lanes 16-31), N = lane&15
  const int rowHalf = (lane >> 4) * 8;
#pragma unroll
  for (int t = 0; t < 4; ++t) {
    const int col = nBase + nw * 64 + t * 16 + (lane & 15);
    const float bv = bias[col];
#pragma unroll
    for (int r = 0; r < 8; ++r) {
      const int row = mBase + mw * 16 + r + rowHalf;
      float v = acc[t][r] + bv;
      if (ACT == 1) v = fmaxf(v, 0.f);
      C[(long)row * N + col] = v;
    }
  }
}

// ---------------------------------------------------------------------------
// K4: attention for one (b, head, 32-query tile).  QK^T and PV both in WMMA.
__global__ __launch_bounds__(256) void attention_kernel(
    const float* __restrict__ qkv, float* __restrict__ out, int S) {
  __shared__ __align__(64) char smem[4096 + 16384 + 32768];
  half_t* Qs = (half_t*)smem;                   // [32][64]
  half_t* Ks = (half_t*)(smem + 4096);          // [128][64] key-major
  float*  Sc = (float*)(smem + 4096 + 16384);   // [32][256]
  half_t* Ps = (half_t*)(smem + 4096);          // [32][256] overlays Ks
  half_t* Vt = (half_t*)(smem + 4096 + 16384);  // [64][256] overlays Sc
  __shared__ float red[32][8];

  const int tid = threadIdx.x, lane = tid & 31, wv = tid >> 5;
  const int qBase = blockIdx.x * 32;
  const int h = blockIdx.y, b = blockIdx.z;
  const long rs = 3 * H_;
  const float scale = 0.125f;   // 1/sqrt(DH)

  for (int e = tid; e < 32 * 64; e += 256) {
    const int q = e >> 6, d = e & 63;
    Qs[q * 64 + d] = (half_t)qkv[(long)(b * S + qBase + q) * rs + h * 64 + d];
  }
  // ---- phase 1: Sc = (Q @ K^T) * scale, keys in 128-wide phases ----
  const int mw1 = wv & 1, nw1 = wv >> 1;        // 2M x 4N, tile 16x32
  for (int kp = 0; kp < S; kp += 128) {
    int nK = S - kp; if (nK > 128) nK = 128;
    __syncthreads();
    for (int e = tid; e < 128 * 64; e += 256) {
      const int key = e >> 6, d = e & 63;
      Ks[key * 64 + d] = (key < nK)
        ? (half_t)qkv[(long)(b * S + kp + key) * rs + H_ + h * 64 + d]
        : (half_t)0.f;
    }
    __syncthreads();
    v8f acc[2] = {};
#pragma unroll
    for (int ks = 0; ks < 2; ++ks) {
      const v16h af = make_a_frag(Qs + (mw1 * 16) * 64, 64, lane, ks * 32);
#pragma unroll
      for (int t = 0; t < 2; ++t) {
        const v16h bf = make_b_frag(Ks, 64, lane, nw1 * 32 + t * 16, ks * 32);
        acc[t] = wmma_f16(af, bf, acc[t]);
      }
    }
    const int rowHalf = (lane >> 4) * 8;
#pragma unroll
    for (int t = 0; t < 2; ++t) {
      const int col = kp + nw1 * 32 + t * 16 + (lane & 15);
#pragma unroll
      for (int r = 0; r < 8; ++r)
        Sc[(mw1 * 16 + r + rowHalf) * 256 + col] = acc[t][r] * scale;
    }
  }
  __syncthreads();
  // ---- softmax rows (8 threads/row) ----
  const int srow = tid >> 3, sub = tid & 7;
  float mx = -3.0e38f;
  for (int c = sub; c < S; c += 8) mx = fmaxf(mx, Sc[srow * 256 + c]);
  red[srow][sub] = mx;
  __syncthreads();
  float mall = red[srow][0];
#pragma unroll
  for (int j = 1; j < 8; ++j) mall = fmaxf(mall, red[srow][j]);
  __syncthreads();
  float se = 0.f;
  for (int c = sub; c < S; c += 8) se += __expf(Sc[srow * 256 + c] - mall);
  red[srow][sub] = se;
  __syncthreads();
  float sall = 0.f;
#pragma unroll
  for (int j = 0; j < 8; ++j) sall += red[srow][j];
  const float inv = 1.f / sall;
  for (int c = sub; c < 256; c += 8) {
    const float p = (c < S) ? __expf(Sc[srow * 256 + c] - mall) * inv : 0.f;
    Ps[srow * 256 + c] = (half_t)p;
  }
  __syncthreads();
  // ---- load V transposed (d-major) over Sc ----
  for (int e = tid; e < 64 * 256; e += 256) {
    const int d = e >> 8, key = e & 255;
    Vt[d * 256 + key] = (key < S)
      ? (half_t)qkv[(long)(b * S + key) * rs + 2 * H_ + h * 64 + d]
      : (half_t)0.f;
  }
  __syncthreads();
  // ---- phase 2: O = P @ V  (M=32, N=64, K=S) ----
  const int mw2 = wv >> 2, nw2 = wv & 3;        // 2M x 4N, tile 16x16
  v8f oacc = {};
  for (int ks = 0; ks < S; ks += 32) {
    const v16h af = make_a_frag(Ps + (mw2 * 16) * 256, 256, lane, ks);
    const v16h bf = make_b_frag(Vt, 256, lane, nw2 * 16, ks);
    oacc = wmma_f16(af, bf, oacc);
  }
  const int rowHalf2 = (lane >> 4) * 8;
  const int col = nw2 * 16 + (lane & 15);
#pragma unroll
  for (int r = 0; r < 8; ++r) {
    const int row = mw2 * 16 + r + rowHalf2;
    out[(long)(b * S + qBase + row) * H_ + h * 64 + col] = oacc[r];
  }
}

// ---------------------------------------------------------------------------
// K5: out = LayerNorm(x + r) * g + b   (one block per row, H=512)
__global__ __launch_bounds__(256) void add_ln_kernel(
    const float* __restrict__ x, const float* __restrict__ r,
    const float* __restrict__ g, const float* __restrict__ bb,
    float* __restrict__ out) {
  __shared__ float s1[256], s2[256];
  const long row = blockIdx.x;
  const int tid = threadIdx.x;
  const float v0 = x[row * H_ + tid] + r[row * H_ + tid];
  const float v1 = x[row * H_ + 256 + tid] + r[row * H_ + 256 + tid];
  s1[tid] = v0 + v1;
  s2[tid] = v0 * v0 + v1 * v1;
  __syncthreads();
  for (int o = 128; o > 0; o >>= 1) {
    if (tid < o) { s1[tid] += s1[tid + o]; s2[tid] += s2[tid + o]; }
    __syncthreads();
  }
  const float mean = s1[0] * (1.f / H_);
  const float var  = s2[0] * (1.f / H_) - mean * mean;
  const float rsq  = rsqrtf(var + 1e-5f);
  out[row * H_ + tid]       = (v0 - mean) * rsq * g[tid] + bb[tid];
  out[row * H_ + 256 + tid] = (v1 - mean) * rsq * g[256 + tid] + bb[256 + tid];
}

// ---------------------------------------------------------------------------
// K6: dense_w [F,V] f32 -> transposed f16 [V][F] (n-major for B fragments)
__global__ __launch_bounds__(256) void convw_kernel(
    const float* __restrict__ w, half_t* __restrict__ wt) {
  const int i = blockIdx.x * 256 + threadIdx.x;   // over V*F
  const int v = i >> 11, f = i & (F_ - 1);
  wt[(long)v * F_ + f] = (half_t)w[(long)f * V_ + v];
}

// ---------------------------------------------------------------------------
// K7: fused joint: for one (b, t, 32-u tile):
//   logits = tanh(a[b,t,:] + l[b,u,:]) @ dense_w + dense_b  -> streaming
//   logsumexp -> blank_lp / label_lp.   M=32, N=512, K=2048.
// B tile (512x32 f16 = 32KB) is DMA'd by the Tensor Data Mover each K-step.
__global__ __launch_bounds__(256) void joint_kernel(
    const float* __restrict__ aproj, const float* __restrict__ lproj,
    const float* __restrict__ db, const half_t* __restrict__ dwt,
    const int* __restrict__ labels, float* __restrict__ blank_lp,
    float* __restrict__ label_lp) {
  __shared__ __align__(64) char jmem[2048 + 32768];
  half_t* As  = (half_t*)jmem;             // [32][32]
  half_t* Bst = (half_t*)(jmem + 2048);    // [512][32] n-major
  float*  chk = (float*)(jmem + 2048);     // [32][256] overlays Bst
  __shared__ float red[32][8];
  __shared__ float rowMax[32], rowSum[32], rowBlank[32], rowTgt[32];

  const int tid = threadIdx.x, lane = tid & 31, wv = tid >> 5;
  const int mw = wv >> 2, nw = wv & 3;     // 2M x 4N: wave tile 16 x 128
  const int t = blockIdx.x, uh = blockIdx.y, b = blockIdx.z;
  const float* aRow  = aproj + (long)(b * T_ + t) * F_;
  const float* lRows = lproj + (long)(b * U_ + uh * 32) * F_;
  const unsigned bstLds = (unsigned)(uintptr_t)Bst;  // LDS byte offset

  v8f acc[8] = {};
  for (int kk = 0; kk < F_; kk += 32) {
    __syncthreads();
    if (wv == 0) {
      // TDM: tile_dim0=32 elems, tile_dim1=512 rows, row stride F_ elems,
      // destination = Bst (row-major tile == n-major/k-contiguous layout).
      tdm_load_tile_f16(bstLds, dwt + kk, F_, V_, 32, V_, F_);
    }
    if (kk + 32 < F_)   // prefetch next activation chunk
      __builtin_prefetch(aRow + kk + 32 + (tid & 31), 0, 1);
    for (int e = tid; e < 1024; e += 256) {       // A tile: tanh(a+l) -> f16
      const int m = e >> 5, c = e & 31;
      As[m * 32 + c] = (half_t)tanhf(aRow[kk + c] + lRows[(long)m * F_ + kk + c]);
    }
    if (wv == 0) __builtin_amdgcn_s_wait_tensorcnt(0);
    __syncthreads();
    const v16h af = make_a_frag(As + (mw * 16) * 32, 32, lane, 0);
#pragma unroll
    for (int tt = 0; tt < 8; ++tt) {
      const v16h bf = make_b_frag(Bst, 32, lane, nw * 128 + tt * 16, 0);
      acc[tt] = wmma_f16(af, bf, acc[tt]);
    }
  }
  // ---- epilogue: streaming logsumexp over two 256-col chunks ----
  const int srow = tid >> 3, sub = tid & 7;
  const int u = uh * 32 + srow;
  const int tg = labels[b * U_ + ((u < U_ - 1) ? (u + 1) : (U_ - 1))];
  const int rowHalf = (lane >> 4) * 8;
  for (int c = 0; c < 2; ++c) {
    __syncthreads();
    if ((nw >> 1) == c) {
#pragma unroll
      for (int tt = 0; tt < 8; ++tt) {
        const int colG = nw * 128 + tt * 16 + (lane & 15);
        const int colL = colG - c * 256;
        const float bv = db[colG];
#pragma unroll
        for (int r = 0; r < 8; ++r)
          chk[(mw * 16 + r + rowHalf) * 256 + colL] = acc[tt][r] + bv;
      }
    }
    __syncthreads();
    float mx = -3.0e38f;
    for (int cc = sub; cc < 256; cc += 8) mx = fmaxf(mx, chk[srow * 256 + cc]);
    red[srow][sub] = mx;
    __syncthreads();
    float mall = red[srow][0];
#pragma unroll
    for (int j = 1; j < 8; ++j) mall = fmaxf(mall, red[srow][j]);
    __syncthreads();
    float se = 0.f;
    for (int cc = sub; cc < 256; cc += 8) se += __expf(chk[srow * 256 + cc] - mall);
    red[srow][sub] = se;
    __syncthreads();
    float sall = 0.f;
#pragma unroll
    for (int j = 0; j < 8; ++j) sall += red[srow][j];
    if (sub == 0) {
      if (c == 0) {
        rowMax[srow] = mall; rowSum[srow] = sall;
        rowBlank[srow] = chk[srow * 256 + 0];
      } else {
        const float om = rowMax[srow], nm = fmaxf(om, mall);
        rowSum[srow] = rowSum[srow] * __expf(om - nm) + sall * __expf(mall - nm);
        rowMax[srow] = nm;
      }
      if (tg >= c * 256 && tg < (c + 1) * 256)
        rowTgt[srow] = chk[srow * 256 + (tg - c * 256)];
    }
    __syncthreads();
  }
  if (sub == 0) {
    const float lse = rowMax[srow] + __logf(rowSum[srow]);
    const long base = ((long)(b * T_ + t)) * U_ + u;
    blank_lp[base] = rowBlank[srow] - lse;
    label_lp[base] = (u < U_ - 1) ? (rowTgt[srow] - lse) : NEGI;
  }
}

// ---------------------------------------------------------------------------
// K8: RNNT alpha DP (serial per batch element; 16K logaddexp each - tiny)
__global__ void rnnt_kernel(const float* __restrict__ blank_lp,
                            const float* __restrict__ label_lp,
                            float* __restrict__ out) {
  __shared__ float losses[B_];
  const int tid = threadIdx.x;
  if (tid < B_) {
    const float* blp = blank_lp + (long)tid * T_ * U_;
    const float* llp = label_lp + (long)tid * T_ * U_;
    float row[U_];
    row[0] = 0.f;
    for (int u = 1; u < U_; ++u) row[u] = row[u - 1] + llp[u - 1];
    for (int t = 1; t < T_; ++t) {
      float prev = NEGI;
      for (int u = 0; u < U_; ++u) {
        const float vb = row[u] + blp[(t - 1) * U_ + u];
        const float vl = prev + ((u > 0) ? llp[t * U_ + (u - 1)] : NEGI);
        const float m = fmaxf(vb, vl);
        const float cur = m + __logf(__expf(vb - m) + __expf(vl - m));
        row[u] = cur; prev = cur;
      }
    }
    losses[tid] = -(row[U_ - 1] + blp[(T_ - 1) * U_ + (U_ - 1)]);
  }
  __syncthreads();
  if (tid == 0)
    out[0] = 0.25f * (losses[0] + losses[1] + losses[2] + losses[3]);
}

// ---------------------------------------------------------------------------
// Host-side orchestration.
// Input order (setup_inputs() insertion order, recursed into params):
//  0 audio_data  1 label_data  2 attention_mask
//  3 audio_pos  4 label_pos  5 word_emb
//  6 + i*12 + j : audio_layers[i] leaf j (insertion order below)
//  54 + i*12 + j: label_layers[i]
//  78 joint_wa  79 joint_ba  80 joint_wl  81 joint_bl  82 dense_w  83 dense_b
enum { W_QKV=0, B_QKV, W_OUT, B_OUT, LN1_G, LN1_B, W1, B1, W2, B2, LN2_G, LN2_B };

extern "C" void kernel_launch(void* const* d_in, const int* in_sizes, int n_in,
                              void* d_out, int out_size, void* d_ws, size_t ws_size,
                              hipStream_t stream) {
  (void)in_sizes; (void)n_in; (void)out_size; (void)ws_size;
  auto gf = [&](int i) { return (const float*)d_in[i]; };
  const float* audio     = gf(0);
  const int*   labels    = (const int*)d_in[1];
  const float* audio_pos = gf(3);
  const float* label_pos = gf(4);
  const float* word_emb  = gf(5);
  const float* joint_wa = gf(78), *joint_ba = gf(79);
  const float* joint_wl = gf(80), *joint_bl = gf(81);
  const float* dense_w  = gf(82), *dense_b  = gf(83);

  float* ws = (float*)d_ws;
  float* XA    = ws + 0;          // [1024,512]
  float* XMID  = ws + 524288;     // [1024,512]
  float* QKV   = ws + 1048576;    // [1024,1536]
  float* AO    = ws + 2621440;    // [1024,512] (shared audio/label)
  float* PR    = ws + 3145728;    // [1024,512]
  float* FFN   = ws + 3670016;    // [1024,2048]
  float* YA    = ws + 5767168;    // [256,512]
  float* YMID  = ws + 5898240;    // [256,512]
  float* QKVL  = ws + 6029312;    // [256,1536]
  float* APROJ = ws + 6422528;    // [1024,2048]
  float* LPROJ = ws + 8519680;    // [256,2048]
  float* BLANK = ws + 9043968;    // [4,256,64]
  float* LABLP = ws + 9109504;    // [4,256,64]
  half_t* DWT  = (half_t*)(ws + 9175040); // [512][2048] f16

  const dim3 blk(256);

  auto run_layer = [&](int base, float* x, float* xmid, float* qkvBuf,
                       int S, int BS) {
    auto L = [&](int leaf) { return gf(base + leaf); };
    gemm_kernel<0><<<dim3(1536 / 128, BS / 64), blk, 0, stream>>>(
        x, L(W_QKV), L(B_QKV), qkvBuf, BS, 1536, 512);
    attention_kernel<<<dim3(S / 32, NH_, B_), blk, 0, stream>>>(qkvBuf, AO, S);
    gemm_kernel<0><<<dim3(512 / 128, BS / 64), blk, 0, stream>>>(
        AO, L(W_OUT), L(B_OUT), PR, BS, 512, 512);
    add_ln_kernel<<<dim3(BS), blk, 0, stream>>>(x, PR, L(LN1_G), L(LN1_B), xmid);
    gemm_kernel<1><<<dim3(2048 / 128, BS / 64), blk, 0, stream>>>(
        xmid, L(W1), L(B1), FFN, BS, 2048, 512);
    gemm_kernel<0><<<dim3(512 / 128, BS / 64), blk, 0, stream>>>(
        FFN, L(W2), L(B2), PR, BS, 512, 2048);
    add_ln_kernel<<<dim3(BS), blk, 0, stream>>>(xmid, PR, L(LN2_G), L(LN2_B), x);
  };

  // audio encoder
  audio_pos_kernel<<<dim3(B_ * T_), blk, 0, stream>>>(audio, audio_pos, XA);
  for (int i = 0; i < 4; ++i) run_layer(6 + i * 12, XA, XMID, QKV, T_, B_ * T_);

  // label encoder
  embed_kernel<<<dim3(B_ * U_), blk, 0, stream>>>(labels, word_emb, label_pos, YA);
  for (int i = 0; i < 2; ++i) run_layer(54 + i * 12, YA, YMID, QKVL, U_, B_ * U_);

  // joint projections
  gemm_kernel<0><<<dim3(2048 / 128, (B_ * T_) / 64), blk, 0, stream>>>(
      XA, joint_wa, joint_ba, APROJ, B_ * T_, 2048, 512);
  gemm_kernel<0><<<dim3(2048 / 128, (B_ * U_) / 64), blk, 0, stream>>>(
      YA, joint_wl, joint_bl, LPROJ, B_ * U_, 2048, 512);

  // fused joint + log-softmax + gather
  convw_kernel<<<dim3((V_ * F_) / 256), blk, 0, stream>>>(dense_w, DWT);
  joint_kernel<<<dim3(T_, 2, B_), blk, 0, stream>>>(
      APROJ, LPROJ, dense_b, DWT, labels, BLANK, LABLP);

  // RNNT DP + mean
  rnnt_kernel<<<dim3(1), dim3(32), 0, stream>>>(BLANK, LABLP, (float*)d_out);
}
